// SNNModelLearnable_56023553409229
// MI455X (gfx1250) — matrix operand order repaired and checked by reading, not compiled, not run on hardware
//
#include <hip/hip_runtime.h>
#include <hip/hip_bf16.h>

typedef __attribute__((ext_vector_type(16))) __bf16 v16bf;
typedef __attribute__((ext_vector_type(8)))  float  v8f;
typedef int v4i __attribute__((vector_size(16)));   // b128 payload type

#define M_BLK 64
#define N_BLK 128
#define K_BLK 32
#define LDSW  40   // padded LDS row stride (bf16 elems) -> 80B, 16B aligned

#define AS1 __attribute__((address_space(1)))
#define AS3 __attribute__((address_space(3)))

union FragA { v16bf v; uint4 u[2]; };
union Pack8 { __bf16 h[8]; uint4 u; };

// ---- CDNA5 async global->LDS copy (ASYNCcnt-tracked, bypasses VGPRs) -------
__device__ __forceinline__ void async_copy_b128(const void* g, void* l) {
#if __has_builtin(__builtin_amdgcn_global_load_async_to_lds_b128)
  __builtin_amdgcn_global_load_async_to_lds_b128(
      (AS1 v4i*)(size_t)g, (AS3 v4i*)(unsigned)(size_t)l, 0, 0);
#else
  asm volatile("global_load_async_to_lds_b128 %0, %1, off"
               :: "v"((unsigned)(size_t)l), "v"(g) : "memory");
#endif
}

__device__ __forceinline__ void wait_async0() {
#if __has_builtin(__builtin_amdgcn_s_wait_asynccnt)
  __builtin_amdgcn_s_wait_asynccnt(0);
#else
  asm volatile("s_wait_asynccnt 0x0" ::: "memory");
#endif
}

// ---------------------------------------------------------------------------
// Fused GEMM (C = A @ W^T + bias) + LIF update, double-buffered LDS.
//   A: [M,K] activations (float for the input layer, bf16 spikes otherwise)
//   W: [N,K] weights bf16
// Block: 128 threads = 4 waves (2x2), wave tile 32x64 -> 8 WMMAs / K-step.
// ---------------------------------------------------------------------------
template <typename TA>
__global__ __launch_bounds__(128) void gemm_lif_kernel(
    const TA* __restrict__ A,
    const __bf16* __restrict__ W,
    const float* __restrict__ bias,
    float* __restrict__ mem,
    __bf16* __restrict__ spk,
    const float* __restrict__ beta_p,
    const float* __restrict__ th_p,
    int M, int N, int K)
{
  __shared__ __bf16 Asmem[2][M_BLK * LDSW];   // 2 x 5120 B
  __shared__ __bf16 Bsmem[2][N_BLK * LDSW];   // 2 x 10240 B

  const int tid  = threadIdx.x;
  const int lane = tid & 31;
  const int wave = tid >> 5;       // 0..3
  const int wm   = wave >> 1;      // 0..1 (M)
  const int wn   = wave & 1;       // 0..1 (N)
  const int l16  = lane & 15;
  const int lhi  = lane >> 4;

  const int block_m = blockIdx.y * M_BLK;
  const int block_n = blockIdx.x * N_BLK;

  const float beta = fminf(fmaxf(beta_p[0], 0.0f), 1.0f);
  const float th   = th_p[0];

  // staging coordinates (128 threads)
  const int arow = tid >> 1;          // 0..63
  const int acol = (tid & 1) * 16;    // 0,16  (16 halfs / thread)
  const int brow = tid;               // 0..127 (32 halfs / thread)

  auto stageA = [&](int buf, int k0) {
    const size_t g = (size_t)(block_m + arow) * K + (size_t)(k0 + acol);
    __bf16* dst = &Asmem[buf][arow * LDSW + acol];
    if constexpr (sizeof(TA) == 4) {
      const float4 f0 = *(const float4*)(A + g);
      const float4 f1 = *(const float4*)(A + g + 4);
      const float4 f2 = *(const float4*)(A + g + 8);
      const float4 f3 = *(const float4*)(A + g + 12);
      Pack8 p0, p1;
      p0.h[0] = (__bf16)f0.x; p0.h[1] = (__bf16)f0.y;
      p0.h[2] = (__bf16)f0.z; p0.h[3] = (__bf16)f0.w;
      p0.h[4] = (__bf16)f1.x; p0.h[5] = (__bf16)f1.y;
      p0.h[6] = (__bf16)f1.z; p0.h[7] = (__bf16)f1.w;
      p1.h[0] = (__bf16)f2.x; p1.h[1] = (__bf16)f2.y;
      p1.h[2] = (__bf16)f2.z; p1.h[3] = (__bf16)f2.w;
      p1.h[4] = (__bf16)f3.x; p1.h[5] = (__bf16)f3.y;
      p1.h[6] = (__bf16)f3.z; p1.h[7] = (__bf16)f3.w;
      *(uint4*)(dst)     = p0.u;
      *(uint4*)(dst + 8) = p1.u;
      if (k0 + K_BLK < K) __builtin_prefetch(A + g + K_BLK, 0, 1);
    } else {
      async_copy_b128(A + g,     dst);
      async_copy_b128(A + g + 8, dst + 8);
    }
  };

  auto stageB = [&](int buf, int k0) {
    const size_t g = (size_t)(block_n + brow) * K + (size_t)k0;
    __bf16* dst = &Bsmem[buf][brow * LDSW];
#pragma unroll
    for (int c = 0; c < 4; ++c)
      async_copy_b128(W + g + c * 8, dst + c * 8);
  };

  v8f acc[2][4] = {};

  // prologue: stage first K tile into buffer 0
  stageA(0, 0);
  stageB(0, 0);
  wait_async0();
  __syncthreads();

  const int nk = K / K_BLK;
  for (int ki = 0; ki < nk; ++ki) {
    const int buf = ki & 1;
    if (ki + 1 < nk) {             // stage next tile into the other buffer
      stageA(buf ^ 1, (ki + 1) * K_BLK);
      stageB(buf ^ 1, (ki + 1) * K_BLK);
    }

    // ---- fragment loads per documented CDNA5 WMMA VGPR layouts
    FragA a[2], b[4];
#pragma unroll
    for (int mi = 0; mi < 2; ++mi) {
      const __bf16* p = &Asmem[buf][(wm * 32 + mi * 16 + l16) * LDSW];
      a[mi].u[0] = *(const uint4*)(p + lhi * 8);        // K = lhi*8 .. +7
      a[mi].u[1] = *(const uint4*)(p + 16 + lhi * 8);   // K = 16+lhi*8 .. +7
    }
#pragma unroll
    for (int ni = 0; ni < 4; ++ni) {
      const __bf16* p = &Bsmem[buf][(wn * 64 + ni * 16 + l16) * LDSW];
      b[ni].u[0] = *(const uint4*)(p + lhi * 8);        // K = lhi*8 .. +7
      b[ni].u[1] = *(const uint4*)(p + 16 + lhi * 8);   // K = 16+lhi*8 .. +7
    }

#pragma unroll
    for (int mi = 0; mi < 2; ++mi)
#pragma unroll
      for (int ni = 0; ni < 4; ++ni)
        acc[mi][ni] = __builtin_amdgcn_wmma_f32_16x16x32_bf16(
            false, a[mi].v, false, b[ni].v, (short)0, acc[mi][ni], false, false);

    wait_async0();     // next tile fully in LDS (no-op on last iteration)
    __syncthreads();   // publish; frag ds_loads already drained by WMMA use
  }

  // ---- fused LIF epilogue (C/D layout: M = r + 8*lhi, N = l16)
#pragma unroll
  for (int mi = 0; mi < 2; ++mi) {
#pragma unroll
    for (int ni = 0; ni < 4; ++ni) {
      const int n = block_n + wn * 64 + ni * 16 + l16;
      const float bv = bias[n];
#pragma unroll
      for (int r = 0; r < 8; ++r) {
        const int m = block_m + wm * 32 + mi * 16 + lhi * 8 + r;
        const size_t idx = (size_t)m * N + n;
        const float cur   = acc[mi][ni][r] + bv;
        const float mo    = mem[idx];
        const float reset = (mo - th > 0.0f) ? 1.0f : 0.0f;
        const float mn    = beta * mo + cur - reset * th;
        mem[idx] = mn;
        spk[idx] = (__bf16)((mn - th > 0.0f) ? 1.0f : 0.0f);
      }
    }
  }
}

// ---------------------------------------------------------------------------
// Last layer: [B,H] spikes @ Wlast[O,H]^T + blast, LIF, write spk/mem to d_out.
// ---------------------------------------------------------------------------
__global__ __launch_bounds__(256) void last_lif_kernel(
    const __bf16* __restrict__ spk_in,   // [B,H]
    const float* __restrict__ Wlast,     // [O,H]
    const float* __restrict__ blast,     // [O]
    float* __restrict__ meml,            // [B,O]
    float* __restrict__ out_spk,         // [B,O] slice of d_out
    float* __restrict__ out_mem,         // [B,O] slice of d_out
    const float* __restrict__ beta_p,
    const float* __restrict__ th_p,
    int H, int O)
{
  __shared__ float red[256];
  const int b   = blockIdx.x;
  const int tid = threadIdx.x;
  const float beta = fminf(fmaxf(beta_p[0], 0.0f), 1.0f);
  const float th   = th_p[0];

  for (int o = 0; o < O; ++o) {
    float s = 0.0f;
    for (int k = tid; k < H; k += 256)
      s += (float)spk_in[(size_t)b * H + k] * Wlast[(size_t)o * H + k];
    red[tid] = s;
    __syncthreads();
    for (int w = 128; w > 0; w >>= 1) {
      if (tid < w) red[tid] += red[tid + w];
      __syncthreads();
    }
    if (tid == 0) {
      const float cur = red[0] + blast[o];
      const size_t idx = (size_t)b * O + o;
      const float mo    = meml[idx];
      const float reset = (mo - th > 0.0f) ? 1.0f : 0.0f;
      const float mn    = beta * mo + cur - reset * th;
      meml[idx]    = mn;
      out_spk[idx] = (mn - th > 0.0f) ? 1.0f : 0.0f;
      out_mem[idx] = mn;
    }
    __syncthreads();
  }
}

// ---------------------------------------------------------------------------
__global__ void f32_to_bf16_kernel(const float* __restrict__ in,
                                   __bf16* __restrict__ out, size_t n)
{
  size_t i = (size_t)blockIdx.x * blockDim.x + threadIdx.x;
  const size_t stride = (size_t)gridDim.x * blockDim.x;
  for (; i < n; i += stride) out[i] = (__bf16)in[i];
}

// ---------------------------------------------------------------------------
extern "C" void kernel_launch(void* const* d_in, const int* in_sizes, int n_in,
                              void* d_out, int out_size, void* d_ws, size_t ws_size,
                              hipStream_t stream) {
  (void)in_sizes; (void)n_in; (void)out_size; (void)ws_size;
  constexpr int T = 64, B = 256, I = 1024, H = 2048, L = 2, O = 10;

  const float* x      = (const float*)d_in[0];
  const float* Wfirst = (const float*)d_in[1];
  const float* bfirst = (const float*)d_in[2];
  const float* Wh     = (const float*)d_in[3];
  const float* bh     = (const float*)d_in[4];
  const float* Wlast  = (const float*)d_in[5];
  const float* blast  = (const float*)d_in[6];
  const float* beta_f = (const float*)d_in[7];
  const float* th_f   = (const float*)d_in[8];
  const float* beta_h = (const float*)d_in[9];
  const float* th_h   = (const float*)d_in[10];
  const float* beta_l = (const float*)d_in[11];
  const float* th_l   = (const float*)d_in[12];

  float* out = (float*)d_out;

  // workspace layout (~28 MB)
  char* ws = (char*)d_ws;
  __bf16* Wf_bf = (__bf16*)ws;  ws += (size_t)H * I * sizeof(__bf16);      // 4 MB
  __bf16* Wh_bf = (__bf16*)ws;  ws += (size_t)L * H * H * sizeof(__bf16);  // 16 MB
  float*  memf  = (float*)ws;   ws += (size_t)B * H * sizeof(float);       // 2 MB
  float*  memh  = (float*)ws;   ws += (size_t)L * B * H * sizeof(float);   // 4 MB
  float*  meml  = (float*)ws;   ws += (size_t)B * O * sizeof(float);
  __bf16* spkA  = (__bf16*)ws;  ws += (size_t)B * H * sizeof(__bf16);      // 1 MB
  __bf16* spkB  = (__bf16*)ws;  ws += (size_t)B * H * sizeof(__bf16);      // 1 MB

  // zero-init membrane states (graph-capture legal)
  (void)hipMemsetAsync(memf, 0, (size_t)B * H * sizeof(float), stream);
  (void)hipMemsetAsync(memh, 0, (size_t)L * B * H * sizeof(float), stream);
  (void)hipMemsetAsync(meml, 0, (size_t)B * O * sizeof(float), stream);

  // weights -> bf16 once per launch (stays resident in the 192 MB L2)
  f32_to_bf16_kernel<<<2048, 256, 0, stream>>>(Wfirst, Wf_bf, (size_t)H * I);
  f32_to_bf16_kernel<<<2048, 256, 0, stream>>>(Wh, Wh_bf, (size_t)L * H * H);

  const dim3 grid(H / N_BLK, B / M_BLK);  // (16, 4)
  for (int t = 0; t < T; ++t) {
    // input layer: f32 x -> bf16 in LDS, WMMA, fused LIF -> spkA
    gemm_lif_kernel<float><<<grid, 128, 0, stream>>>(
        x + (size_t)t * B * I, Wf_bf, bfirst, memf, spkA, beta_f, th_f, B, H, I);
    // hidden layer 0: spkA -> spkB
    gemm_lif_kernel<__bf16><<<grid, 128, 0, stream>>>(
        spkA, Wh_bf, bh, memh, spkB, beta_h, th_h, B, H, H);
    // hidden layer 1: spkB -> spkA
    gemm_lif_kernel<__bf16><<<grid, 128, 0, stream>>>(
        spkB, Wh_bf + (size_t)H * H, bh + H, memh + (size_t)B * H, spkA,
        beta_h + 1, th_h + 1, B, H, H);
    // output layer + LIF, writes spk_rec[t] and mem_rec[t]
    last_lif_kernel<<<B, 256, 0, stream>>>(
        spkA, Wlast, blast, meml,
        out + (size_t)t * B * O,
        out + (size_t)T * B * O + (size_t)t * B * O,
        beta_l, th_l, H, O);
  }
}
